// LOIMLossSafe_79654463471714
// MI455X (gfx1250) — matrix operand order repaired and checked by reading, not compiled, not run on hardware
//
#include <hip/hip_runtime.h>

// ---------------------------------------------------------------------------
// OIM loss for MI455X (gfx1250, wave32).
//  - logits GEMM on v_wmma_f32_16x16x32_bf16
//  - A row-tiles staged into LDS with global_load_async_to_lds_b128
//    (ASYNCcnt), double-buffered so DMA overlaps the WMMA chain
//  - all 8 A fragments preloaded into VGPRs so ds_load latency hides
//    under the WMMA chain (partial dscnt waits instead of full stalls)
//  - B (lut/cq) streamed from HBM once, converted bf16 into LDS
//  - softmax fused with fixed shift 30 (logits bounded in [-30,30])
//  - sequential memory-bank update parallelized per-label / per-queue-slot
// ---------------------------------------------------------------------------

#define DIM   256
#define NPID  100000
#define NCQ   50000
#define NROW  1024
#define SCAL  30.0f
#define EPSN  1e-12f
#define BCOLS 64          // lut/cq columns per GEMM block
#define BSTR  272         // padded LDS row stride (bf16 elements)

typedef __attribute__((ext_vector_type(16))) __bf16       v16bf;
typedef __attribute__((ext_vector_type(8)))  float        v8f;
typedef __attribute__((ext_vector_type(8)))  unsigned int v8u;

__device__ __forceinline__ unsigned short f2bf(float f) {
    unsigned u = __float_as_uint(f);
    u += 0x7FFFu + ((u >> 16) & 1u);          // round-to-nearest-even
    return (unsigned short)(u >> 16);
}

__device__ __forceinline__ v16bf mk16(uint4 lo, uint4 hi) {
    union { v8u u; v16bf b; } c;
    c.u = (v8u){lo.x, lo.y, lo.z, lo.w, hi.x, hi.y, hi.z, hi.w};
    return c.b;
}

// ---- zero per-row softmax accumulators -----------------------------------
__global__ void k_init(float* rowsum, float* pick, float* badp) {
    int i = blockIdx.x * blockDim.x + threadIdx.x;
    if (i < NROW) { rowsum[i] = 0.f; pick[i] = 0.f; badp[i] = 0.f; }
}

// ---- L2-normalize inputs; emit fp32 + bf16 copies (one wave per row) -----
__global__ void k_norm(const float* __restrict__ in, float* __restrict__ xf,
                       unsigned short* __restrict__ xb) {
    int w = (blockIdx.x * blockDim.x + threadIdx.x) >> 5;
    int lane = threadIdx.x & 31;
    if (w >= NROW) return;
    const float* s = in + (size_t)w * DIM + lane * 8;
    float4 a = *(const float4*)(s);
    float4 b = *(const float4*)(s + 4);
    float ss = a.x*a.x + a.y*a.y + a.z*a.z + a.w*a.w
             + b.x*b.x + b.y*b.y + b.z*b.z + b.w*b.w;
    #pragma unroll
    for (int m = 1; m < 32; m <<= 1) ss += __shfl_xor(ss, m, 32);
    float inv = 1.0f / fmaxf(sqrtf(ss), EPSN);
    a.x *= inv; a.y *= inv; a.z *= inv; a.w *= inv;
    b.x *= inv; b.y *= inv; b.z *= inv; b.w *= inv;
    *(float4*)(xf + (size_t)w * DIM + lane * 8)     = a;
    *(float4*)(xf + (size_t)w * DIM + lane * 8 + 4) = b;
    uint4 p;
    p.x = (unsigned)f2bf(a.x) | ((unsigned)f2bf(a.y) << 16);
    p.y = (unsigned)f2bf(a.z) | ((unsigned)f2bf(a.w) << 16);
    p.z = (unsigned)f2bf(b.x) | ((unsigned)f2bf(b.y) << 16);
    p.w = (unsigned)f2bf(b.z) | ((unsigned)f2bf(b.w) << 16);
    *(uint4*)(xb + (size_t)w * DIM + lane * 8) = p;
}

// ---- mean(ious) < 0.2 gate ------------------------------------------------
__global__ void k_cond(const float* __restrict__ ious, int* cond) {
    __shared__ float sm[256];
    float s = 0.f;
    for (int i = threadIdx.x; i < NROW; i += 256) s += ious[i];
    sm[threadIdx.x] = s; __syncthreads();
    for (int o = 128; o > 0; o >>= 1) {
        if (threadIdx.x < o) sm[threadIdx.x] += sm[threadIdx.x + o];
        __syncthreads();
    }
    if (threadIdx.x == 0) *cond = (sm[0] * (1.0f / NROW) < 0.2f) ? 1 : 0;
}

// ---- copy lut/cq into outputs (out base is d_out+1 -> scalar stores) -----
__global__ void k_copy(const float* __restrict__ lut, const float* __restrict__ cq,
                       float* __restrict__ olut, float* __restrict__ ocq) {
    const size_t lim    = (size_t)NPID * (DIM / 4);
    const size_t total4 = ((size_t)NPID + NCQ) * (DIM / 4);
    for (size_t i = blockIdx.x * (size_t)blockDim.x + threadIdx.x;
         i < total4; i += (size_t)gridDim.x * blockDim.x) {
        float4 v; float* d;
        if (i < lim) {
            __builtin_prefetch(((const float4*)lut) + i + 2048, 0, 0);
            v = ((const float4*)lut)[i];      d = olut + i * 4;
        } else {
            v = ((const float4*)cq)[i - lim]; d = ocq + (i - lim) * 4;
        }
        d[0] = v.x; d[1] = v.y; d[2] = v.z; d[3] = v.w;
    }
}

// ---- fused bf16 WMMA GEMM + online exp-sum --------------------------------
// block = 128 threads (4 waves); each wave owns a 16-column strip.
// B tile (64 cols x 256 K) -> bf16 in LDS, 8 fragments hoisted to VGPRs.
// A row-tiles (16x256 bf16) DMA'd into a double-buffered LDS region with
// global_load_async_to_lds_b128 so the copy of tile rt+1 overlaps the
// 8x v_wmma_f32_16x16x32_bf16 chain + softmax epilogue of tile rt.
__global__ __launch_bounds__(128)
void k_gemm(const float* __restrict__ lut, const float* __restrict__ cq,
            const unsigned short* __restrict__ xb, float* __restrict__ rowsum) {
    __shared__ unsigned short Bsh[BCOLS * BSTR];
    __shared__ unsigned short Ash[2 * 16 * BSTR];   // double-buffered A tile
    __shared__ int nzp[BCOLS][2];
    __shared__ int flg[BCOLS];

    const int t = threadIdx.x;
    const int wv = t >> 5, lane = t & 31;

    // --- kick off async DMA of A tile 0 into buffer 0 ---------------------
    // one instruction moves one 512B row (32 lanes x 16B); wave w owns rows
    // 4w..4w+3 of the tile.
    {
        const int r0 = wv * 4;
        #pragma unroll
        for (int rr = 0; rr < 4; ++rr) {
            const unsigned short* gsrc = xb + (size_t)(r0 + rr) * DIM + lane * 8;
            unsigned dst = (unsigned)(size_t)&Ash[(r0 + rr) * BSTR + lane * 8];
            asm volatile("global_load_async_to_lds_b128 %0, %1, off"
                         :: "v"(dst), "v"(gsrc) : "memory");
        }
    }

    // --- stage B: 2 threads per column, each converts 128 fp32 -> bf16 ----
    {
        const int c = t >> 1, h = t & 1;
        const long g = (long)blockIdx.x * BCOLS + c;
        const float* src = nullptr; int pad = 0;
        if (g < NPID)            src = lut + (size_t)g * DIM;
        else if (g < NPID + NCQ) src = cq  + (size_t)(g - NPID) * DIM;
        else                     pad = 1;
        int nz = 0;
        const int kb = h * 128;
        for (int k = 0; k < 128; k += 4) {
            float4 v = pad ? make_float4(0.f, 0.f, 0.f, 0.f)
                           : *(const float4*)(src + kb + k);
            nz |= (v.x != 0.f) | (v.y != 0.f) | (v.z != 0.f) | (v.w != 0.f);
            uint2 p;
            p.x = (unsigned)f2bf(v.x) | ((unsigned)f2bf(v.y) << 16);
            p.y = (unsigned)f2bf(v.z) | ((unsigned)f2bf(v.w) << 16);
            *(uint2*)&Bsh[c * BSTR + kb + k] = p;
        }
        nzp[c][h] = nz;
        __syncthreads();
        if (h == 0) flg[c] = pad ? 2 : ((nzp[c][0] | nzp[c][1]) ? 0 : 1);
        asm volatile("s_wait_asynccnt 0x0" ::: "memory");   // tile 0 landed
        __syncthreads();
    }

    const int l15 = lane & 15, half = lane >> 4;
    const int col = wv * 16 + l15;

    // hoist all 8 B fragments (K=256) into registers for this wave's strip.
    // 16-bit B 32x16 layout: lanes 0-15 K {0..7,16..23}, lanes 16-31 +8.
    v16bf bfr[8];
    #pragma unroll
    for (int kk = 0; kk < 8; ++kk) {
        const unsigned short* bp = &Bsh[col * BSTR + kk * 32 + half * 8];
        bfr[kk] = mk16(*(const uint4*)bp, *(const uint4*)(bp + 16));
    }
    const int fl = flg[col];

    for (int rt = 0; rt < NROW / 16; ++rt) {
        // prefetch tile rt+1 into the other buffer (overlaps WMMA below);
        // safe: everyone finished reading that buffer (tile rt-1) before the
        // barrier at the end of the previous iteration.
        const int nxt = rt + 1;
        if (nxt < NROW / 16) {
            const int r0 = wv * 4, b16 = (nxt & 1) * 16;
            #pragma unroll
            for (int rr = 0; rr < 4; ++rr) {
                const unsigned short* gsrc =
                    xb + (size_t)(nxt * 16 + r0 + rr) * DIM + lane * 8;
                unsigned dst =
                    (unsigned)(size_t)&Ash[(b16 + r0 + rr) * BSTR + lane * 8];
                asm volatile("global_load_async_to_lds_b128 %0, %1, off"
                             :: "v"(dst), "v"(gsrc) : "memory");
            }
        }

        // 16-bit A 16x32 layout: lanes 0-15 K {0..7,16..23}, lanes 16-31 +8.
        // Preload ALL 8 fragments into distinct VGPRs first: the 16
        // ds_load_b128 issue back-to-back and their latency hides under the
        // WMMA chain (partial dscnt waits instead of s_wait_dscnt 0).
        const unsigned short* abase =
            &Ash[(rt & 1) * 16 * BSTR + l15 * BSTR + half * 8];
        v16bf afr[8];
        #pragma unroll
        for (int kk = 0; kk < 8; ++kk) {
            uint4 lo = *(const uint4*)(abase + kk * 32);
            uint4 hi = *(const uint4*)(abase + kk * 32 + 16);
            afr[kk] = mk16(lo, hi);
        }

        v8f acc = {0.f, 0.f, 0.f, 0.f, 0.f, 0.f, 0.f, 0.f};
        #pragma unroll
        for (int kk = 0; kk < 8; ++kk) {
            acc = __builtin_amdgcn_wmma_f32_16x16x32_bf16(
                false, afr[kk], false, bfr[kk], (short)0, acc, false, false);
        }

        // softmax partial: fixed shift 30 (all logits in [-30,30])
        float e[8];
        #pragma unroll
        for (int r = 0; r < 8; ++r) {
            float logit = (fl == 1) ? -1.f : acc[r];   // empty row -> -1
            float p = __expf(logit * SCAL - SCAL);
            e[r] = (fl == 2) ? 0.f : p;                // grid padding -> 0
        }
        #pragma unroll
        for (int m = 1; m < 16; m <<= 1) {             // reduce over 16 cols
            #pragma unroll
            for (int r = 0; r < 8; ++r) e[r] += __shfl_xor(e[r], m, 32);
        }
        if (l15 == 0) {
            const int rb = rt * 16 + half * 8;
            #pragma unroll
            for (int r = 0; r < 8; ++r) atomicAdd(&rowsum[rb + r], e[r]);
        }

        // publish tile rt+1: my DMAs done, then block-wide barrier.
        asm volatile("s_wait_asynccnt 0x0" ::: "memory");
        __syncthreads();
    }
}

// ---- fp32 picked logit + bad-prototype flag (one wave per row) -----------
__global__ void k_pick(const float* __restrict__ lut, const int* __restrict__ label,
                       const float* __restrict__ xf, float* __restrict__ pick,
                       float* __restrict__ badp) {
    int w = (blockIdx.x * blockDim.x + threadIdx.x) >> 5;
    int lane = threadIdx.x & 31;
    if (w >= NROW) return;
    int y = label[w];
    if (y >= NPID) { if (lane == 0) { pick[w] = 0.f; badp[w] = 0.f; } return; }
    const float* lr = lut + (size_t)y * DIM;
    const float* xr = xf + (size_t)w * DIM;
    float dot = 0.f; int nz = 0;
    for (int k = lane; k < DIM; k += 32) {
        float lv = lr[k]; nz |= (lv != 0.f); dot += lv * xr[k];
    }
    #pragma unroll
    for (int m = 1; m < 32; m <<= 1) {
        dot += __shfl_xor(dot, m, 32); nz |= __shfl_xor(nz, m, 32);
    }
    if (lane == 0) {
        int bad = (nz == 0);
        pick[w] = bad ? 1.f : dot;      // bad own prototype -> logit 1
        badp[w] = bad ? 1.f : 0.f;
    }
}

// ---- final CE reduction ---------------------------------------------------
__global__ void k_loss(const float* __restrict__ rowsum, const float* __restrict__ pick,
                       const float* __restrict__ badp, const int* __restrict__ label,
                       float* __restrict__ out) {
    __shared__ float sm[256];
    float s = 0.f;
    for (int i = threadIdx.x; i < NROW; i += 256) {
        if (label[i] < NPID) {
            // GEMM counted the (masked, -1) self column; swap exp(-60)->exp(0)
            float corr = badp[i] * (1.f - __expf(-2.f * SCAL));
            float lse = SCAL + logf(rowsum[i] + corr);
            s += lse - SCAL * pick[i];
        }
    }
    sm[threadIdx.x] = s; __syncthreads();
    for (int o = 128; o > 0; o >>= 1) {
        if (threadIdx.x < o) sm[threadIdx.x] += sm[threadIdx.x + o];
        __syncthreads();
    }
    if (threadIdx.x == 0) out[0] = sm[0] * (1.0f / NROW);
}

// ---- lut update: one wave replays the full chain of its label ------------
__global__ void k_upd_lut(const int* __restrict__ label, const float* __restrict__ ious,
                          const float* __restrict__ xf, float* __restrict__ olut,
                          const int* __restrict__ cond) {
    if (*cond == 0) return;
    int w = (blockIdx.x * blockDim.x + threadIdx.x) >> 5;
    int lane = threadIdx.x & 31;
    if (w >= NROW) return;
    int y = label[w];
    if (y >= NPID) return;
    int prev = 0;
    for (int j = lane; j < w; j += 32) prev |= (label[j] == y);
    if (__any(prev)) return;            // only first occurrence owns the chain
    float* dst = olut + (size_t)y * DIM;
    float r[8];
    #pragma unroll
    for (int q = 0; q < 8; ++q) r[q] = dst[lane * 8 + q];
    for (int pass = 0; pass < 2; ++pass) {          // pass1 momentum, pass2 iou
        for (int j = w; j < NROW; ++j) {
            if (label[j] != y) continue;
            float bc = pass ? ious[j] : 0.5f;
            float ac = 1.f - bc;
            const float* xr = xf + (size_t)j * DIM + lane * 8;
            float ss = 0.f;
            #pragma unroll
            for (int q = 0; q < 8; ++q) { r[q] = ac * r[q] + bc * xr[q]; ss += r[q] * r[q]; }
            #pragma unroll
            for (int m = 1; m < 32; m <<= 1) ss += __shfl_xor(ss, m, 32);
            float inv = 1.f / fmaxf(sqrtf(ss), EPSN);
            #pragma unroll
            for (int q = 0; q < 8; ++q) r[q] *= inv;
        }
    }
    #pragma unroll
    for (int q = 0; q < 8; ++q) dst[lane * 8 + q] = r[q];
}

// ---- cq update: slots are disjoint (2U <= 2048 < 50000) -> parallel ------
__global__ void k_upd_cq(const int* __restrict__ label, const float* __restrict__ xf,
                         const int* __restrict__ header, float* __restrict__ ocq,
                         const int* __restrict__ cond) {
    if (*cond == 0) return;
    int w = (blockIdx.x * blockDim.x + threadIdx.x) >> 5;
    int lane = threadIdx.x & 31;
    if (w >= NROW) return;
    if (label[w] < NPID) return;
    int rank = 0, U = 0;
    for (int j = lane; j < NROW; j += 32) {
        int u = (label[j] >= NPID);
        U += u;
        if (j < w) rank += u;
    }
    #pragma unroll
    for (int m = 1; m < 32; m <<= 1) { U += __shfl_xor(U, m, 32); rank += __shfl_xor(rank, m, 32); }
    long head = (long)header[0];
    long s1 = (head + rank) % NCQ;          // first pass write
    long s2 = (head + U + rank) % NCQ;      // second pass write
    const float* xr = xf + (size_t)w * DIM;
    float* d1 = ocq + (size_t)s1 * DIM;
    float* d2 = ocq + (size_t)s2 * DIM;
    for (int k = lane; k < DIM; k += 32) { float v = xr[k]; d1[k] = v; d2[k] = v; }
}

extern "C" void kernel_launch(void* const* d_in, const int* in_sizes, int n_in,
                              void* d_out, int out_size, void* d_ws, size_t ws_size,
                              hipStream_t stream) {
    const float* inputs = (const float*)d_in[0];
    const int*   label  = (const int*)d_in[1];
    const float* ious   = (const float*)d_in[2];
    const float* lut    = (const float*)d_in[3];
    const float* cq     = (const float*)d_in[4];
    const int*   header = (const int*)d_in[5];

    float* out      = (float*)d_out;
    float* out_loss = out;
    float* out_lut  = out + 1;
    float* out_cq   = out + 1 + (size_t)NPID * DIM;

    char* ws = (char*)d_ws;
    float*          xf     = (float*)ws;                                   // 1 MB
    unsigned short* xb     = (unsigned short*)(ws + 1048576);              // 512 KB
    float*          rowsum = (float*)(ws + 1048576 + 524288);              // 4 KB
    float*          pick   = rowsum + NROW;
    float*          badp   = pick + NROW;
    int*            cond   = (int*)(badp + NROW);

    k_init<<<4, 256, 0, stream>>>(rowsum, pick, badp);
    k_norm<<<128, 256, 0, stream>>>(inputs, xf, xb);
    k_cond<<<1, 256, 0, stream>>>(ious, cond);
    k_copy<<<4096, 256, 0, stream>>>(lut, cq, out_lut, out_cq);

    int gblocks = (NPID + NCQ + BCOLS - 1) / BCOLS;
    k_gemm<<<gblocks, 128, 0, stream>>>(lut, cq, xb, rowsum);

    k_pick<<<128, 256, 0, stream>>>(lut, label, xf, pick, badp);
    k_loss<<<1, 256, 0, stream>>>(rowsum, pick, badp, label, out_loss);
    k_upd_lut<<<128, 256, 0, stream>>>(label, ious, xf, out_lut, cond);
    k_upd_cq<<<128, 256, 0, stream>>>(label, xf, header, out_cq, cond);
}
